// PagedLinear_7370163879939
// MI455X (gfx1250) — matrix operand order repaired and checked by reading, not compiled, not run on hardware
//
#include <hip/hip_runtime.h>

// ---------------------------------------------------------------------------
// PagedLinear forward_topk for MI455X (gfx1250, wave32, WMMA).
//   logits = x @ W^T + bias   (N=4096, D=2048, V=50257), top-64 per row.
//
// Pipeline:
//   1) f32 -> bf16 pre-pass for x and W into workspace (bandwidth bound,
//      ~30us; removes ALL conversion VALU from the GEMM hot loop and halves
//      operand bytes per MAC).
//   2) WMMA GEMM, register blocked 2 M-tiles x 8 N-tiles per wave
//      (32x128 output, 128 acc VGPRs, 16 v_wmma per k-step). Fragments are
//      raw global_load_b128 -> bit_cast. global_prefetch_b8 streams W.
//      Per-chunk (1024 cols) sorted top-64 per row via wave32 shuffle argmax,
//      run in two 16-row phases through one 64KB LDS tile.
//   3) 50-way sorted-list merge per row -> final (vals, idx).
// ---------------------------------------------------------------------------

#define N_ROWS 4096
#define DDIM   2048
#define VDIM   50257
#define TOPK   64

#define TM     32                            // rows per block (2 WMMA M tiles)
#define MT     2                             // M tiles per wave
#define WAVES  8                             // 256 threads
#define CHUNK  1024                          // V columns per block
#define PASSC  128                           // WAVES * 16 columns
#define NPASS  (CHUNK / PASSC)               // 8 column-tiles per wave
#define KSTEP  32                            // WMMA K
#define NCHUNK ((VDIM + CHUNK - 1) / CHUNK)  // 50 chunks over V

#define NEG_INF (-3.402823466e38f)

typedef __bf16 bf16_t;
typedef bf16_t       v16bf __attribute__((ext_vector_type(16)));
typedef float        v8f   __attribute__((ext_vector_type(8)));
typedef unsigned int v8u   __attribute__((ext_vector_type(8)));

// f32 -> bf16 RNE (pre-pass only; not on the GEMM hot path).
__device__ __forceinline__ unsigned int f2bf(float f) {
    unsigned u = __float_as_uint(f);
    u += 0x7FFFu + ((u >> 16) & 1u);
    return u >> 16;
}

// ---------------------------------------------------------------------------
// Pre-pass: convert f32 buffer to packed bf16 (8 elements / thread).
// ---------------------------------------------------------------------------
__global__ __launch_bounds__(256)
void cvt_f32_to_bf16(const float* __restrict__ in,
                     unsigned int* __restrict__ out, long long n) {
    long long base = ((long long)blockIdx.x * 256 + threadIdx.x) * 8;
    if (base + 8 > n) return;
    const float4* p = (const float4*)(in + base);
    float4 a = p[0], b = p[1];
    uint4 o;
    o.x = f2bf(a.x) | (f2bf(a.y) << 16);
    o.y = f2bf(a.z) | (f2bf(a.w) << 16);
    o.z = f2bf(b.x) | (f2bf(b.y) << 16);
    o.w = f2bf(b.z) | (f2bf(b.w) << 16);
    *(uint4*)(out + (base >> 1)) = o;
}

// ---------------------------------------------------------------------------
// Fragment loads: pure b128 loads + bit_cast, no conversion ALU.
// A (16x32 bf16, ISA 7.12.2): lanes 0-15: M=lane, K={k0..+7, k0+16..+23};
// lanes 16-31: M=lane-16, K={k0+8..+15, k0+24..+31}.
// ---------------------------------------------------------------------------
__device__ __forceinline__ v16bf load_a_frag(const unsigned short* __restrict__ row,
                                             int k0, int half) {
    uint4 lo = *(const uint4*)(row + k0 + half * 8);        // 8 bf16
    uint4 hi = *(const uint4*)(row + k0 + 16 + half * 8);   // 8 bf16
    v8u t; t[0]=lo.x; t[1]=lo.y; t[2]=lo.z; t[3]=lo.w;
           t[4]=hi.x; t[5]=hi.y; t[6]=hi.z; t[7]=hi.w;
    return __builtin_bit_cast(v16bf, t);
}

// B (32x16 bf16): lane = column; lanes 0-15 hold K={k0..+7, k0+16..+23},
// lanes 16-31 hold K={k0+8..+15, k0+24..+31} (same K order as A).
__device__ __forceinline__ v16bf load_b_frag(const unsigned short* __restrict__ row,
                                             int k0, int half) {
    uint4 lo = *(const uint4*)(row + k0 + half * 8);
    uint4 hi = *(const uint4*)(row + k0 + 16 + half * 8);
    v8u t; t[0]=lo.x; t[1]=lo.y; t[2]=lo.z; t[3]=lo.w;
           t[4]=hi.x; t[5]=hi.y; t[6]=hi.z; t[7]=hi.w;
    return __builtin_bit_cast(v16bf, t);
}

// ---------------------------------------------------------------------------
// Kernel: one block = 32 rows x 1024 V-columns of logits.
// ---------------------------------------------------------------------------
__global__ __launch_bounds__(256)
void pagedlinear_gemm_topk_chunk(const unsigned short* __restrict__ xb,
                                 const unsigned short* __restrict__ Wb,
                                 const float* __restrict__ bias,
                                 float* __restrict__ cand_val,
                                 int*   __restrict__ cand_idx) {
    __shared__ float s_logits[16 * CHUNK];   // 64 KB, reused for both phases

    const int lane  = threadIdx.x & 31;
    const int wv    = threadIdx.x >> 5;
    const int half  = lane >> 4;
    const int m     = lane & 15;
    const int n0    = blockIdx.x * TM;
    const int c     = blockIdx.y;
    const int vbase = c * CHUNK;

    const unsigned short* xrow[MT];
#pragma unroll
    for (int t = 0; t < MT; ++t)
        xrow[t] = xb + (size_t)(n0 + t * 16 + m) * DDIM;

    const unsigned short* wrow[NPASS];
#pragma unroll
    for (int p = 0; p < NPASS; ++p) {
        int v = vbase + p * PASSC + wv * 16 + m;   // lane's B column
        if (v >= VDIM) v = VDIM - 1;               // clamp (masked at store)
        wrow[p] = Wb + (size_t)v * DDIM;
    }

    v8f acc[MT][NPASS];
#pragma unroll
    for (int t = 0; t < MT; ++t)
#pragma unroll
        for (int p = 0; p < NPASS; ++p)
            acc[t][p] = (v8f){0.f, 0.f, 0.f, 0.f, 0.f, 0.f, 0.f, 0.f};

    for (int k0 = 0; k0 < DDIM; k0 += KSTEP) {
        v16bf a0 = load_a_frag(xrow[0], k0, half);
        v16bf a1 = load_a_frag(xrow[1], k0, half);
#pragma unroll
        for (int p = 0; p < NPASS; ++p) {
            __builtin_prefetch(wrow[p] + k0 + 2 * KSTEP, 0, 1);
            v16bf b = load_b_frag(wrow[p], k0, half);
            acc[0][p] = __builtin_amdgcn_wmma_f32_16x16x32_bf16(
                false, a0, false, b, (short)0, acc[0][p], false, false);
            acc[1][p] = __builtin_amdgcn_wmma_f32_16x16x32_bf16(
                false, a1, false, b, (short)0, acc[1][p], false, false);
        }
    }

    // Two phases of 16 rows each through the 64KB LDS tile.
#pragma unroll
    for (int t = 0; t < MT; ++t) {
        if (t) __syncthreads();   // phase-1 selection done before overwrite

        // C/D layout: lanes 0-15: N=lane, M=i; lanes 16-31: N=lane-16, M=8+i.
#pragma unroll
        for (int p = 0; p < NPASS; ++p) {
            int colc = p * PASSC + wv * 16 + m;
            int gcol = vbase + colc;
            float b = (gcol < VDIM) ? bias[gcol] : 0.f;
#pragma unroll
            for (int i = 0; i < 8; ++i) {
                int row   = half * 8 + i;
                float val = (gcol < VDIM) ? (acc[t][p][i] + b) : NEG_INF;
                s_logits[row * CHUNK + colc] = val;
            }
        }
        __syncthreads();

        // Per-row top-64 of the chunk: 2 rows per wave; each lane caches its
        // 32 strided values in VGPRs; 64 rounds of local-max -> wave32
        // xor-butterfly argmax -> knock out the winner.
#pragma unroll
        for (int rr = 0; rr < 2; ++rr) {
            const int r = wv * 2 + rr;
            const int n = n0 + t * 16 + r;
            float lv[CHUNK / 32];
#pragma unroll
            for (int j = 0; j < CHUNK / 32; ++j)
                lv[j] = s_logits[r * CHUNK + j * 32 + lane];

            for (int it = 0; it < TOPK; ++it) {
                float bv = NEG_INF; int bj = 0;
#pragma unroll
                for (int j = 0; j < CHUNK / 32; ++j)
                    if (lv[j] > bv) { bv = lv[j]; bj = j; }

                float v  = bv;
                int   gi = vbase + bj * 32 + lane;
#pragma unroll
                for (int off = 16; off > 0; off >>= 1) {
                    float ov = __shfl_xor(v,  off, 32);
                    int   oi = __shfl_xor(gi, off, 32);
                    if (ov > v || (ov == v && oi < gi)) { v = ov; gi = oi; }
                }
                const int wcol  = gi - vbase;
                const int wlane = wcol & 31;
                const int wj    = wcol >> 5;
#pragma unroll
                for (int j = 0; j < CHUNK / 32; ++j)   // keep lv in VGPRs
                    if (lane == wlane && j == wj) lv[j] = NEG_INF;

                if (lane == 0) {
                    size_t o = ((size_t)n * NCHUNK + c) * TOPK + it;
                    cand_val[o] = v;
                    cand_idx[o] = gi;
                }
            }
        }
    }
}

// ---------------------------------------------------------------------------
// 50-way sorted-list merge per row (each chunk's 64 candidates descending).
// One wave per row; lanes own <=2 chunk head pointers; 64 argmax rounds.
// ---------------------------------------------------------------------------
__global__ __launch_bounds__(64)
void pagedlinear_merge_topk(const float* __restrict__ cand_val,
                            const int*   __restrict__ cand_idx,
                            float* __restrict__ out_val,
                            float* __restrict__ out_idx) {
    __shared__ float s_val[2 * NCHUNK * TOPK];
    __shared__ int   s_idx[2 * NCHUNK * TOPK];

    const int lane = threadIdx.x & 31;
    const int wv   = threadIdx.x >> 5;
    const int n    = blockIdx.x * 2 + wv;
    const int base = wv * NCHUNK * TOPK;

    for (int t = lane; t < NCHUNK * TOPK; t += 32) {
        size_t o = (size_t)n * NCHUNK * TOPK + t;
        s_val[base + t] = cand_val[o];
        s_idx[base + t] = cand_idx[o];
    }
    __syncthreads();

    int cA = lane, cB = lane + 32;
    int hA = 0,    hB = 0;

    for (int it = 0; it < TOPK; ++it) {
        float va = (hA < TOPK) ? s_val[base + cA * TOPK + hA] : NEG_INF;
        float vb = (cB < NCHUNK && hB < TOPK) ? s_val[base + cB * TOPK + hB]
                                              : NEG_INF;
        float v; int ch, hp;
        if (vb > va) { v = vb; ch = cB; hp = hB; }
        else         { v = va; ch = cA; hp = hA; }

#pragma unroll
        for (int off = 16; off > 0; off >>= 1) {
            float ov  = __shfl_xor(v,  off, 32);
            int   och = __shfl_xor(ch, off, 32);
            int   ohp = __shfl_xor(hp, off, 32);
            if (ov > v || (ov == v && och < ch)) { v = ov; ch = och; hp = ohp; }
        }
        if (ch < 32) { if (lane == ch)      hA++; }
        else         { if (lane == ch - 32) hB++; }

        if (lane == 0) {
            out_val[(size_t)n * TOPK + it] = v;
            out_idx[(size_t)n * TOPK + it] =
                (float)s_idx[base + ch * TOPK + hp];   // idx < 2^24: exact
        }
    }
}

// ---------------------------------------------------------------------------
extern "C" void kernel_launch(void* const* d_in, const int* in_sizes, int n_in,
                              void* d_out, int out_size, void* d_ws,
                              size_t ws_size, hipStream_t stream) {
    (void)in_sizes; (void)n_in; (void)out_size; (void)ws_size;

    const float* x    = (const float*)d_in[0];   // [N, D] f32
    const float* W    = (const float*)d_in[1];   // [V, D] f32
    const float* bias = (const float*)d_in[2];   // [V]    f32

    float* out_val = (float*)d_out;
    float* out_idx = (float*)d_out + (size_t)N_ROWS * TOPK;

    // Workspace layout (bytes):
    //   [0)                 x_bf16 : N*D*2   = 16,777,216
    //   [xb_end)            W_bf16 : V*D*2   = 205,852,672
    //   [wb_end)            cand_val: N*NCHUNK*64*4 = 52,428,800
    //   [cv_end)            cand_idx: same          = 52,428,800
    char* ws = (char*)d_ws;
    unsigned short* xb = (unsigned short*)ws;
    unsigned short* Wb = (unsigned short*)(ws + (size_t)N_ROWS * DDIM * 2);
    float* cand_val = (float*)(ws + (size_t)N_ROWS * DDIM * 2
                                  + (size_t)VDIM * DDIM * 2);
    int*   cand_idx = (int*)((char*)cand_val
                             + (size_t)N_ROWS * NCHUNK * TOPK * sizeof(float));

    const long long nx = (long long)N_ROWS * DDIM;   // 8,388,608  (mult of 8)
    const long long nw = (long long)VDIM * DDIM;     // 102,926,336 (mult of 8)

    cvt_f32_to_bf16<<<(unsigned)(nx / (256 * 8)), 256, 0, stream>>>(
        x, (unsigned int*)xb, nx);
    cvt_f32_to_bf16<<<(unsigned)(nw / (256 * 8)), 256, 0, stream>>>(
        W, (unsigned int*)Wb, nw);

    dim3 g1(N_ROWS / TM, NCHUNK);
    pagedlinear_gemm_topk_chunk<<<g1, 256, 0, stream>>>(
        xb, Wb, bias, cand_val, cand_idx);

    pagedlinear_merge_topk<<<N_ROWS / 2, 64, 0, stream>>>(
        cand_val, cand_idx, out_val, out_idx);
}
